// Quantification_75033078661554
// MI455X (gfx1250) — compile-verified, matrix operand order checked by reading, not run on hardware
//
#include <hip/hip_runtime.h>

// ---------------------------------------------------------------------------
// Problem constants (from reference): B=65536 rows, DIM=256, K=1024 codes.
// Outputs concatenated flat: quantize [B*DIM] f32, diff [1] f32, embed_ind [B]
// (stored as float). out_size = 16777216 + 1 + 65536.
// ---------------------------------------------------------------------------
constexpr int BROWS = 65536;
constexpr int DIM   = 256;
constexpr int KCB   = 1024;
constexpr int NTILES = KCB / 16;

typedef __attribute__((ext_vector_type(16))) __bf16 bf16x16;
typedef __attribute__((ext_vector_type(2)))  __bf16 bf16x2;
typedef __attribute__((ext_vector_type(8)))  float  v8f;

// Workspace layout (bytes)
constexpr size_t WS_ESQ   = 0;                                   // 1024 f32
constexpr size_t WS_ETF32 = 4096;                                // [K][D] f32 (1 MB)
constexpr size_t WS_ETHI  = WS_ETF32 + (size_t)KCB * DIM * 4;    // [K][D] bf16 (512 KB)
constexpr size_t WS_ETLO  = WS_ETHI  + (size_t)KCB * DIM * 2;    // [K][D] bf16 (512 KB)
constexpr size_t WS_IDX   = WS_ETLO  + (size_t)KCB * DIM * 2;    // [B] i32 (256 KB)
constexpr size_t WS_ROWS  = WS_IDX   + (size_t)BROWS * 4;        // [B] f32 (256 KB)

// K index inside a 32-wide chunk for operand VGPR pair v (0..7), lane-half h.
__device__ __forceinline__ int wmma_k(int v, int h) {
    return ((v & 4) ? (16 + ((v & 3) << 1)) : (v << 1)) + (h << 3);
}

// ---------------------------------------------------------------------------
// Kernel 1: transpose embed [D][K] -> [K][D] as fp32 + split-bf16 hi/lo, and
// compute e_sq[k] = sum_d embed[d][k]^2 exactly in fp32.
// grid = K, block = DIM
// ---------------------------------------------------------------------------
__global__ void vq_prep_embed(const float* __restrict__ embed,
                              float* __restrict__ esq,
                              float* __restrict__ eTf32,
                              __bf16* __restrict__ eThi,
                              __bf16* __restrict__ eTlo) {
    const int k = blockIdx.x;
    const int d = threadIdx.x;
    const float v = embed[(size_t)d * KCB + k];
    eTf32[(size_t)k * DIM + d] = v;
    const __bf16 h = (__bf16)v;
    eThi[(size_t)k * DIM + d] = h;
    eTlo[(size_t)k * DIM + d] = (__bf16)(v - (float)h);

    __shared__ float s[DIM];
    s[d] = v * v;
    __syncthreads();
    for (int o = DIM / 2; o > 0; o >>= 1) {
        if (d < o) s[d] += s[d + o];
        __syncthreads();
    }
    if (d == 0) esq[k] = s[0];
}

// ---------------------------------------------------------------------------
// Kernel 2: distance GEMM + running argmin.
// grid = 512 workgroups x 256 threads (8 waves, wave32). Each wave owns 16
// rows of x (A tile, split-bf16, resident in VGPRs) and sweeps all 64 N-tiles
// of 16 codes. B tiles are double-buffered: tile nt+1 is fetched from global
// into staging registers while tile nt (in LDS, shared by all 8 waves) feeds
// 24 v_wmma_f32_16x16x32_bf16 (split-bf16: hi*hi + hi*lo + lo*hi per K-chunk).
// ---------------------------------------------------------------------------
__global__ __launch_bounds__(256, 1)
void vq_dist_argmin(const float* __restrict__ x,
                    const __bf16* __restrict__ eThi,
                    const __bf16* __restrict__ eTlo,
                    const float* __restrict__ esq,
                    int* __restrict__ widx,
                    float* __restrict__ oidx) {
    __shared__ unsigned int ldsb[4096];  // 8 KB hi tile + 8 KB lo tile

    const int tid   = threadIdx.x;
    const int wave  = tid >> 5;
    const int lane  = tid & 31;
    const int laneq = lane & 15;   // N column within tile (B/C layout)
    const int half  = lane >> 4;   // lane half selects K-group
    const int waveRow = blockIdx.x * 128 + wave * 16;
    const int row     = waveRow + laneq;   // A: M = lane&15 in both halves

    // ---- Load + split A tile into registers: 8 chunks of K=32.
    // Layout puts (k,k+1) in VGPR v and (k+2,k+3) in VGPR v+1, so one float4
    // load fills two consecutive VGPR pairs (4 vector elements).
    bf16x16 ahi[8], alo[8];
    const float* xrow = x + (size_t)row * DIM;
#pragma unroll
    for (int c = 0; c < 8; ++c) {
#pragma unroll
        for (int g = 0; g < 4; ++g) {
            const int kbase = c * 32 + (half << 3) + ((g & 1) << 2) + ((g >> 1) << 4);
            const float4 xv = *(const float4*)(xrow + kbase);
            const int e = ((g >> 1) << 3) + ((g & 1) << 2);  // vector element base
            const __bf16 h0 = (__bf16)xv.x, h1 = (__bf16)xv.y;
            const __bf16 h2 = (__bf16)xv.z, h3 = (__bf16)xv.w;
            ahi[c][e]     = h0; ahi[c][e + 1] = h1;
            ahi[c][e + 2] = h2; ahi[c][e + 3] = h3;
            alo[c][e]     = (__bf16)(xv.x - (float)h0);
            alo[c][e + 1] = (__bf16)(xv.y - (float)h1);
            alo[c][e + 2] = (__bf16)(xv.z - (float)h2);
            alo[c][e + 3] = (__bf16)(xv.w - (float)h3);
        }
    }

    float best[8];
    int   bidx[8];
#pragma unroll
    for (int r = 0; r < 8; ++r) { best[r] = 3.4e38f; bidx[r] = 0; }

    uint4* dsth = (uint4*)ldsb;
    uint4* dstl = (uint4*)(ldsb + 2048);

    // ---- Prologue: fetch tile 0 into staging registers ----
    uint4 rh0, rh1, rl0, rl1;
    {
        const uint4* srch = (const uint4*)eThi;
        const uint4* srcl = (const uint4*)eTlo;
        rh0 = srch[tid]; rh1 = srch[tid + 256];
        rl0 = srcl[tid]; rl1 = srcl[tid + 256];
    }

    for (int nt = 0; nt < NTILES; ++nt) {
        // ---- Commit staged tile nt to LDS ----
        dsth[tid]       = rh0;
        dsth[tid + 256] = rh1;
        dstl[tid]       = rl0;
        dstl[tid + 256] = rl1;
        __syncthreads();

        // ---- Kick off global fetch of tile nt+1 (wait lands at next store) ----
        if (nt + 1 < NTILES) {
            const uint4* srch = (const uint4*)(eThi + (size_t)(nt + 1) * 16 * DIM);
            const uint4* srcl = (const uint4*)(eTlo + (size_t)(nt + 1) * 16 * DIM);
            rh0 = srch[tid]; rh1 = srch[tid + 256];
            rl0 = srcl[tid]; rl1 = srcl[tid + 256];
        }

        const int   nglob = nt * 16 + laneq;
        const float eq    = esq[nglob];   // L2-hot, overlapped with WMMAs

        v8f acc = {};  // f32 accumulator for x . e over full DIM
#pragma unroll
        for (int c = 0; c < 8; ++c) {
            bf16x16 bh, bl;
#pragma unroll
            for (int v = 0; v < 8; ++v) {
                const int k = c * 32 + wmma_k(v, half);
                const unsigned int ui = laneq * 128 + (k >> 1);
                const bf16x2 ph = __builtin_bit_cast(bf16x2, ldsb[ui]);
                const bf16x2 pl = __builtin_bit_cast(bf16x2, ldsb[2048 + ui]);
                bh[2 * v] = ph.x; bh[2 * v + 1] = ph.y;
                bl[2 * v] = pl.x; bl[2 * v + 1] = pl.y;
            }
            // split-bf16: hi*hi + hi*lo + lo*hi ~ fp32-accurate dot product
            acc = __builtin_amdgcn_wmma_f32_16x16x32_bf16(false, ahi[c], false, bh,
                                                          (short)0, acc, false, false);
            acc = __builtin_amdgcn_wmma_f32_16x16x32_bf16(false, ahi[c], false, bl,
                                                          (short)0, acc, false, false);
            acc = __builtin_amdgcn_wmma_f32_16x16x32_bf16(false, alo[c], false, bh,
                                                          (short)0, acc, false, false);
        }

        // dist (up to the row-constant ||x||^2) = ||e_n||^2 - 2 * dot
#pragma unroll
        for (int r = 0; r < 8; ++r) {
            const float dd = fmaf(-2.0f, acc[r], eq);
            if (dd < best[r]) { best[r] = dd; bidx[r] = nglob; }
        }
        __syncthreads();
    }

    // ---- Cross-lane argmin: C tile stripes N across 16 lanes per half ----
#pragma unroll
    for (int r = 0; r < 8; ++r) {
        float d = best[r];
        int   i = bidx[r];
#pragma unroll
        for (int m = 8; m >= 1; m >>= 1) {
            const float od = __shfl_xor(d, m, 32);
            const int   oi = __shfl_xor(i, m, 32);
            if (od < d || (od == d && oi < i)) { d = od; i = oi; }
        }
        if (laneq == 0) {
            const int orow = waveRow + r + (half << 3);  // M = r (+8 in hi half)
            widx[orow] = i;
            oidx[orow] = (float)i;
        }
    }
}

// ---------------------------------------------------------------------------
// Kernel 3: gather winning fp32 code row (quantize == quantize_st numerically)
// and per-row squared error. grid = B, block = DIM.
// ---------------------------------------------------------------------------
__global__ void vq_gather(const float* __restrict__ x,
                          const float* __restrict__ eTf32,
                          const int* __restrict__ widx,
                          float* __restrict__ outq,
                          float* __restrict__ rowsum) {
    const int b = blockIdx.x;
    const int d = threadIdx.x;
    const int i = widx[b];
    const float q  = eTf32[(size_t)i * DIM + d];
    const float xv = x[(size_t)b * DIM + d];
    outq[(size_t)b * DIM + d] = q;
    const float t = q - xv;

    __shared__ float s[DIM];
    s[d] = t * t;
    __syncthreads();
    for (int o = DIM / 2; o > 0; o >>= 1) {
        if (d < o) s[d] += s[d + o];
        __syncthreads();
    }
    if (d == 0) rowsum[b] = s[0];
}

// ---------------------------------------------------------------------------
// Kernel 4: deterministic final reduction of B row sums -> mean. 1 block.
// ---------------------------------------------------------------------------
__global__ void vq_reduce_diff(const float* __restrict__ rowsum,
                               float* __restrict__ diff) {
    const int t = threadIdx.x;
    float s = 0.0f;
    for (int j = 0; j < BROWS / 256; ++j) s += rowsum[(size_t)j * 256 + t];
    __shared__ float sm[256];
    sm[t] = s;
    __syncthreads();
    for (int o = 128; o > 0; o >>= 1) {
        if (t < o) sm[t] += sm[t + o];
        __syncthreads();
    }
    if (t == 0) *diff = sm[0] * (1.0f / ((float)BROWS * (float)DIM));
}

// ---------------------------------------------------------------------------
extern "C" void kernel_launch(void* const* d_in, const int* in_sizes, int n_in,
                              void* d_out, int out_size, void* d_ws, size_t ws_size,
                              hipStream_t stream) {
    (void)in_sizes; (void)n_in; (void)out_size; (void)ws_size;
    const float* x     = (const float*)d_in[0];
    const float* embed = (const float*)d_in[1];

    char* wsb = (char*)d_ws;
    float*  esq    = (float*)(wsb + WS_ESQ);
    float*  eTf32  = (float*)(wsb + WS_ETF32);
    __bf16* eThi   = (__bf16*)(wsb + WS_ETHI);
    __bf16* eTlo   = (__bf16*)(wsb + WS_ETLO);
    int*    widx   = (int*)(wsb + WS_IDX);
    float*  rowsum = (float*)(wsb + WS_ROWS);

    float* outq = (float*)d_out;                          // [B*DIM]
    float* diff = outq + (size_t)BROWS * DIM;             // [1]
    float* oidx = diff + 1;                               // [B] indices as float

    vq_prep_embed<<<dim3(KCB), dim3(DIM), 0, stream>>>(embed, esq, eTf32, eThi, eTlo);
    vq_dist_argmin<<<dim3(BROWS / 128), dim3(256), 0, stream>>>(x, eThi, eTlo, esq,
                                                                widx, oidx);
    vq_gather<<<dim3(BROWS), dim3(DIM), 0, stream>>>(x, eTf32, widx, outq, rowsum);
    vq_reduce_diff<<<dim3(1), dim3(256), 0, stream>>>(rowsum, diff);
}
